// RopeCausalSelfAttention_80522046865831
// MI455X (gfx1250) — matrix operand order repaired and checked by reading, not compiled, not run on hardware
//
#include <hip/hip_runtime.h>
#include <hip/hip_bf16.h>

typedef __bf16 bf16;
typedef __attribute__((ext_vector_type(8)))  __bf16 v8bf;
typedef __attribute__((ext_vector_type(16))) __bf16 v16bf;
typedef __attribute__((ext_vector_type(8)))  float  v8f;
typedef unsigned int u32x4 __attribute__((ext_vector_type(4)));
typedef int          i32x4 __attribute__((ext_vector_type(4)));
typedef int          i32x8 __attribute__((ext_vector_type(8)));

#if defined(__has_builtin)
#if __has_builtin(__builtin_amdgcn_tensor_load_to_lds)
#define USE_TDM 1
#endif
#endif
#ifndef USE_TDM
#define USE_TDM 0
#endif

static __device__ __forceinline__ v16bf cat8(v8bf a, v8bf b) {
    return __builtin_shufflevector(a, b, 0,1,2,3,4,5,6,7,8,9,10,11,12,13,14,15);
}

static __device__ __forceinline__ v8f wmma_bf16(v16bf a, v16bf b, v8f c) {
    return __builtin_amdgcn_wmma_f32_16x16x32_bf16(false, a, false, b, (short)0, c, false, false);
}

#if USE_TDM
// 2D TDM tile load: tile (tile_rows x tile_k) bf16 from row-major [rows][K]
// into LDS with a 16B pad appended after every 128B row (72-elem LDS stride).
static __device__ __forceinline__ void tdm_load_2d(unsigned lds_addr,
                                                   const void* gptr,
                                                   unsigned tensor_k, unsigned tensor_rows,
                                                   unsigned tile_k, unsigned tile_rows,
                                                   unsigned stride_k) {
    unsigned long long ga = (unsigned long long)gptr;
    u32x4 g0;
    g0[0] = 1u;                                  // count=1 (valid user descriptor)
    g0[1] = lds_addr;                            // LDS byte address
    g0[2] = (unsigned)ga;                        // global_addr[31:0]
    g0[3] = (unsigned)(ga >> 32) | (2u << 30);   // global_addr[56:32] | type=2
    i32x8 g1;
    g1[0] = (int)((1u << 16)  |                  // data_size = 2 bytes
                  (1u << 20)  |                  // pad_enable
                  (4u << 22)  |                  // pad_interval = 32 DWORDs (128B)
                  (3u << 25));                   // pad_amount   = 4 DWORDs (16B)
    g1[1] = (int)((tensor_k & 0xFFFFu) << 16);   // tensor_dim0[15:0] @ bit 48
    g1[2] = (int)((tensor_k >> 16) | ((tensor_rows & 0xFFFFu) << 16)); // dim0 hi | dim1 lo
    g1[3] = (int)((tensor_rows >> 16) | (tile_k << 16));               // dim1 hi | tile_dim0
    g1[4] = (int)(tile_rows & 0xFFFFu);          // tile_dim1 (tile_dim2 = 0)
    g1[5] = (int)stride_k;                       // tensor_dim0_stride[31:0]
    g1[6] = 0;                                   // stride0[47:32] | stride1[15:0]
    g1[7] = 0;
    i32x4 z4 = {0, 0, 0, 0};
#if __clang_major__ >= 23
    i32x8 z8 = {0, 0, 0, 0, 0, 0, 0, 0};
    __builtin_amdgcn_tensor_load_to_lds(g0, g1, z4, z4, z8, 0);
#else
    __builtin_amdgcn_tensor_load_to_lds(g0, g1, z4, z4, 0);
#endif
}
#endif

// ---------------------------------------------------------------------------
// fp32 -> bf16 elementwise conversion
// ---------------------------------------------------------------------------
__global__ __launch_bounds__(256) void cvt_f32_bf16(const float* __restrict__ in,
                                                    bf16* __restrict__ out, int n) {
    int i = blockIdx.x * 256 + threadIdx.x;
    if (i < n) out[i] = (bf16)in[i];
}

// ---------------------------------------------------------------------------
// fp32 [K][N] -> bf16 [N][K] transpose (weights), 32x32 tiles via LDS
// ---------------------------------------------------------------------------
__global__ __launch_bounds__(256) void transpose_f32_bf16(const float* __restrict__ in,
                                                          bf16* __restrict__ out,
                                                          int K, int N) {
    __shared__ float t[32][33];
    int tx = threadIdx.x & 31, tg = threadIdx.x >> 5;   // 8 groups of 32
    int k0 = blockIdx.x * 32, n0 = blockIdx.y * 32;
#pragma unroll
    for (int i = 0; i < 4; ++i) {
        int ty = tg * 4 + i;
        t[ty][tx] = in[(size_t)(k0 + ty) * N + n0 + tx];
    }
    __syncthreads();
#pragma unroll
    for (int i = 0; i < 4; ++i) {
        int ty = tg * 4 + i;
        out[(size_t)(n0 + ty) * K + k0 + tx] = (bf16)t[tx][ty];
    }
}

// ---------------------------------------------------------------------------
// bf16 GEMM:  C[M,N] = A[M,K] * Bt[N,K]^T + bias[N]
// Both operands row-major with contiguous K. 256 threads / 8 waves.
// Block tile 128x128, BK=64 (2 WMMA k-steps), double-buffered LDS staged by
// the Tensor Data Mover (fallback: 16B vector copies).
// Wave (wm 0..3, wn 0..1) computes 32x64 = 2x4 WMMA accumulators.
// ---------------------------------------------------------------------------
template <bool OUT_F32>
__global__ __launch_bounds__(256) void gemm_bf16(const bf16* __restrict__ A,
                                                 const bf16* __restrict__ Bt,
                                                 const float* __restrict__ bias,
                                                 void* __restrict__ Cout,
                                                 int M, int N, int K) {
    constexpr int BM = 128, BN = 128, BK = 64;
    constexpr int LD = BK + 8;                     // 72 elems = 144B rows (16B aligned)
    __shared__ __align__(16) bf16 sA[2][BM * LD];
    __shared__ __align__(16) bf16 sB[2][BN * LD];

    const int tid  = threadIdx.x;
    const int wave = tid >> 5, lane = tid & 31;
    const int half = lane >> 4, lr = lane & 15;
    const int wm = wave >> 1, wn = wave & 1;
    const int m0 = blockIdx.x * BM, n0 = blockIdx.y * BN;

    v8f acc[2][4] = {};

    auto stage = [&](int buf, int k0) {
#if USE_TDM
        if (wave == 0) {
            tdm_load_2d((unsigned)(uintptr_t)&sA[buf][0],
                        A + (size_t)m0 * K + k0, K, M, BK, BM, K);
            tdm_load_2d((unsigned)(uintptr_t)&sB[buf][0],
                        Bt + (size_t)n0 * K + k0, K, N, BK, BN, K);
        }
#else
#pragma unroll
        for (int i = 0; i < 4; ++i) {
            int c = tid + i * 256;                 // 1024 16B chunks per operand
            int row = c >> 3, cc = c & 7;
            *(v8bf*)(sA[buf] + row * LD + cc * 8) =
                *(const v8bf*)(A + (size_t)(m0 + row) * K + k0 + cc * 8);
            *(v8bf*)(sB[buf] + row * LD + cc * 8) =
                *(const v8bf*)(Bt + (size_t)(n0 + row) * K + k0 + cc * 8);
        }
#endif
    };

    stage(0, 0);
    const int nIter = K / BK;
    for (int it = 0; it < nIter; ++it) {
        const int buf = it & 1;
#if USE_TDM
        if (wave == 0) __builtin_amdgcn_s_wait_tensorcnt(0);
#endif
        __syncthreads();                            // tile `it` ready; buf^1 free
        if (it + 1 < nIter) stage(buf ^ 1, (it + 1) * BK);

        const bf16* bufA = sA[buf];
        const bf16* bufB = sB[buf];
#pragma unroll
        for (int ks = 0; ks < BK; ks += 32) {
            v16bf af[2], bfm[4];
#pragma unroll
            for (int mi = 0; mi < 2; ++mi) {
                const bf16* p = bufA + (wm * 32 + mi * 16 + lr) * LD + ks + half * 8;
                af[mi] = cat8(*(const v8bf*)p, *(const v8bf*)(p + 16));
            }
#pragma unroll
            for (int ni = 0; ni < 4; ++ni) {
                const bf16* p = bufB + (wn * 64 + ni * 16 + lr) * LD + ks + half * 8;
                bfm[ni] = cat8(*(const v8bf*)p, *(const v8bf*)(p + 16));
            }
#pragma unroll
            for (int mi = 0; mi < 2; ++mi)
#pragma unroll
                for (int ni = 0; ni < 4; ++ni)
                    acc[mi][ni] = wmma_bf16(af[mi], bfm[ni], acc[mi][ni]);
        }
    }

    // epilogue: C layout -> row = r + 8*half, col = lane&15
#pragma unroll
    for (int mi = 0; mi < 2; ++mi)
#pragma unroll
        for (int ni = 0; ni < 4; ++ni) {
            int col = n0 + wn * 64 + ni * 16 + lr;
            float bv = bias[col];
#pragma unroll
            for (int r = 0; r < 8; ++r) {
                int row = m0 + wm * 32 + mi * 16 + r + 8 * half;
                float v = acc[mi][ni][r] + bv;
                if constexpr (OUT_F32)
                    ((float*)Cout)[(size_t)row * N + col] = v;
                else
                    ((bf16*)Cout)[(size_t)row * N + col] = (bf16)v;
            }
        }
}

// ---------------------------------------------------------------------------
// RoPE + head scatter.
// qkv[B,S,3E] bf16 -> q[B,H,S,D], k[B,H,S,D] (rotated), vt[B,H,D,S]
// ---------------------------------------------------------------------------
__global__ __launch_bounds__(256) void rope_scatter(const bf16* __restrict__ qkv,
                                                    bf16* __restrict__ q,
                                                    bf16* __restrict__ k,
                                                    bf16* __restrict__ vt,
                                                    int B, int S, int H, int D) {
    const int d2 = D / 2;
    int idx = blockIdx.x * 256 + threadIdx.x;
    int total = B * S * H * d2;
    if (idx >= total) return;
    int j = idx % d2;
    int h = (idx / d2) % H;
    int s = (idx / (d2 * H)) % S;
    int b = idx / (d2 * H * S);
    const int E = H * D;

    const bf16* row = qkv + (size_t)(b * S + s) * (3 * E);
    float q1 = (float)row[h * D + j],         q2 = (float)row[h * D + j + d2];
    float k1 = (float)row[E + h * D + j],     k2 = (float)row[E + h * D + j + d2];
    float v1 = (float)row[2 * E + h * D + j], v2 = (float)row[2 * E + h * D + j + d2];

    float f   = exp2f(-((float)(2 * j) / (float)D) * 13.287712379549449f); // 10000^(-2j/D)
    float ang = (float)s * f;
    float sn = __sinf(ang), cs = __cosf(ang);

    size_t base = (((size_t)b * H + h) * S + s) * (size_t)D;
    q[base + j]      = (bf16)(q1 * cs - q2 * sn);
    q[base + j + d2] = (bf16)(q2 * cs + q1 * sn);
    k[base + j]      = (bf16)(k1 * cs - k2 * sn);
    k[base + j + d2] = (bf16)(k2 * cs + k1 * sn);

    size_t vb = ((size_t)b * H + h) * (size_t)D * S;
    vt[vb + (size_t)j * S + s]        = (bf16)v1;
    vt[vb + (size_t)(j + d2) * S + s] = (bf16)v2;
}

// ---------------------------------------------------------------------------
// Flash attention (causal, online softmax). One wave per 32-row Q tile:
// K/V B-fragments are reused across both 16-row M tiles (2x WMMA per load).
// block = 128 threads (4 independent waves). grid = (B*H, S/128).
// Q,K: [B,H,S,D] bf16 row-major; Vt: [B,H,D,S] bf16; O: [B,S,E] bf16.
// ---------------------------------------------------------------------------
__global__ __launch_bounds__(128) void flash_attn(const bf16* __restrict__ Q,
                                                  const bf16* __restrict__ Kt,
                                                  const bf16* __restrict__ Vt,
                                                  bf16* __restrict__ O,
                                                  int B, int S, int H, int D) {
    const int wave = threadIdx.x >> 5, lane = threadIdx.x & 31;
    const int half = lane >> 4, lr = lane & 15;
    const int bh = blockIdx.x;
    const int b = bh / H, h = bh % H;
    const int qBase = (blockIdx.y * 4 + wave) * 32;
    const int E = H * D;

    const bf16* Qp = Q  + (size_t)bh * S * D;
    const bf16* Kp = Kt + (size_t)bh * S * D;
    const bf16* Vp = Vt + (size_t)bh * D * S;

    __shared__ __align__(16) bf16 sP[4][32 * 32];
    bf16* myP = sP[wave];

    // Q A-fragments: 2 M tiles x 2 d-steps
    v16bf qf[2][2];
#pragma unroll
    for (int mi = 0; mi < 2; ++mi)
#pragma unroll
        for (int t = 0; t < 2; ++t) {
            const bf16* p = Qp + (size_t)(qBase + mi * 16 + lr) * D + t * 32 + half * 8;
            qf[mi][t] = cat8(*(const v8bf*)p, *(const v8bf*)(p + 16));
        }

    v8f acc[2][4] = {};
    float mrow[2][8], lsum[2][8];
#pragma unroll
    for (int mi = 0; mi < 2; ++mi)
#pragma unroll
        for (int r = 0; r < 8; ++r) { mrow[mi][r] = -INFINITY; lsum[mi][r] = 0.f; }

    const float scale = 0.125f * 1.4426950408889634f;   // 1/sqrt(64) * log2(e)

    for (int kb = 0; kb < qBase + 32; kb += 32) {
        // ---- S = Q K^T : K fragment reused across both M tiles
        v8f sc[2][2] = {};
#pragma unroll
        for (int nt = 0; nt < 2; ++nt)
#pragma unroll
            for (int t = 0; t < 2; ++t) {
                const bf16* p = Kp + (size_t)(kb + nt * 16 + lr) * D + t * 32 + half * 8;
                v16bf kf = cat8(*(const v8bf*)p, *(const v8bf*)(p + 16));
#pragma unroll
                for (int mi = 0; mi < 2; ++mi)
                    sc[mi][nt] = wmma_bf16(qf[mi][t], kf, sc[mi][nt]);
            }

        // ---- causal mask + online softmax, then stage P into LDS
        float alpha[2][8];
#pragma unroll
        for (int mi = 0; mi < 2; ++mi) {
            float sv[2][8];
#pragma unroll
            for (int nt = 0; nt < 2; ++nt) {
                int key = kb + nt * 16 + lr;
#pragma unroll
                for (int r = 0; r < 8; ++r) {
                    int row = qBase + mi * 16 + r + 8 * half;
                    sv[nt][r] = (key <= row) ? sc[mi][nt][r] * scale : -INFINITY;
                }
            }
#pragma unroll
            for (int r = 0; r < 8; ++r) {
                float mr = fmaxf(sv[0][r], sv[1][r]);
                mr = fmaxf(mr, __shfl_xor(mr, 1));
                mr = fmaxf(mr, __shfl_xor(mr, 2));
                mr = fmaxf(mr, __shfl_xor(mr, 4));
                mr = fmaxf(mr, __shfl_xor(mr, 8));
                float mn = fmaxf(mrow[mi][r], mr);
                alpha[mi][r] = exp2f(mrow[mi][r] - mn);
                mrow[mi][r] = mn;
            }
#pragma unroll
            for (int nt = 0; nt < 2; ++nt)
#pragma unroll
                for (int r = 0; r < 8; ++r) {
                    float p = exp2f(sv[nt][r] - mrow[mi][r]);   // 0 when masked
                    sv[nt][r] = p;
                    myP[(mi * 16 + r + 8 * half) * 32 + nt * 16 + lr] = (bf16)p;
                }
#pragma unroll
            for (int r = 0; r < 8; ++r) {
                float ls = sv[0][r] + sv[1][r];
                ls += __shfl_xor(ls, 1);
                ls += __shfl_xor(ls, 2);
                ls += __shfl_xor(ls, 4);
                ls += __shfl_xor(ls, 8);
                lsum[mi][r] = lsum[mi][r] * alpha[mi][r] + ls;
            }
        }
        asm volatile("s_wait_dscnt 0" ::: "memory");    // wave-local LDS RAW

        v16bf pf[2];
#pragma unroll
        for (int mi = 0; mi < 2; ++mi) {
            const bf16* pp = myP + (mi * 16 + lr) * 32 + half * 8;
            pf[mi] = cat8(*(const v8bf*)pp, *(const v8bf*)(pp + 16));
        }

        // ---- rescale accumulators, then acc += P * V (V frag reused over mi)
#pragma unroll
        for (int mi = 0; mi < 2; ++mi)
#pragma unroll
            for (int dt = 0; dt < 4; ++dt)
#pragma unroll
                for (int r = 0; r < 8; ++r)
                    acc[mi][dt][r] *= alpha[mi][r];
#pragma unroll
        for (int dt = 0; dt < 4; ++dt) {
            const bf16* vp = Vp + (size_t)(dt * 16 + lr) * S + kb + half * 8;
            v16bf vf = cat8(*(const v8bf*)vp, *(const v8bf*)(vp + 16));
#pragma unroll
            for (int mi = 0; mi < 2; ++mi)
                acc[mi][dt] = wmma_bf16(pf[mi], vf, acc[mi][dt]);
        }
    }

    // ---- epilogue: O[b, qBase+row, h*D + d] = acc / lsum
#pragma unroll
    for (int mi = 0; mi < 2; ++mi)
#pragma unroll
        for (int dt = 0; dt < 4; ++dt) {
            int d = dt * 16 + lr;
#pragma unroll
            for (int r = 0; r < 8; ++r) {
                int row = qBase + mi * 16 + r + 8 * half;
                float o = acc[mi][dt][r] / lsum[mi][r];
                O[((size_t)b * S + row) * E + h * D + d] = (bf16)o;
            }
        }
}

// ---------------------------------------------------------------------------
extern "C" void kernel_launch(void* const* d_in, const int* in_sizes, int n_in,
                              void* d_out, int out_size, void* d_ws, size_t ws_size,
                              hipStream_t stream) {
    const float* x      = (const float*)d_in[0];
    const float* w_qkv  = (const float*)d_in[1];
    const float* b_qkv  = (const float*)d_in[2];
    const float* w_proj = (const float*)d_in[3];
    const float* b_proj = (const float*)d_in[4];

    const int B = 4, S = 2048, E = 1024, H = 16, D = 64;
    const int M = B * S;            // 8192
    const int N3 = 3 * E;           // 3072

    size_t off = 0;
    auto alloc = [&](size_t bytes) -> void* {
        void* p = (char*)d_ws + off;
        off += (bytes + 255) & ~(size_t)255;
        return p;
    };
    bf16* xb    = (bf16*)alloc((size_t)M * E * 2);
    bf16* wqT   = (bf16*)alloc((size_t)E * N3 * 2);      // [3E][E]
    bf16* wpT   = (bf16*)alloc((size_t)E * E * 2);       // [E][E]
    bf16* qkvb  = (bf16*)alloc((size_t)M * N3 * 2);
    bf16* qh    = (bf16*)alloc((size_t)B * H * S * D * 2);
    bf16* kh    = (bf16*)alloc((size_t)B * H * S * D * 2);
    bf16* vth   = (bf16*)alloc((size_t)B * H * D * S * 2);
    bf16* attn  = (bf16*)alloc((size_t)M * E * 2);

    // input conversion + weight transposes (one tiny extra pass over 8MB)
    cvt_f32_bf16<<<(M * E + 255) / 256, 256, 0, stream>>>(x, xb, M * E);
    transpose_f32_bf16<<<dim3(E / 32, N3 / 32), 256, 0, stream>>>(w_qkv, wqT, E, N3);
    transpose_f32_bf16<<<dim3(E / 32, E / 32), 256, 0, stream>>>(w_proj, wpT, E, E);

    // 1) qkv = x @ w_qkv + b_qkv    (bf16 out)
    gemm_bf16<false><<<dim3(M / 128, N3 / 128), 256, 0, stream>>>(
        xb, wqT, b_qkv, qkvb, M, N3, E);

    // 2) RoPE + head scatter (+ V transpose)
    {
        int total = B * S * H * (D / 2);
        rope_scatter<<<(total + 255) / 256, 256, 0, stream>>>(
            qkvb, qh, kh, vth, B, S, H, D);
    }

    // 3) flash attention
    flash_attn<<<dim3(B * H, S / 128), 128, 0, stream>>>(qh, kh, vth, attn, B, S, H, D);

    // 4) out = attn @ w_proj + b_proj   (fp32 out)
    gemm_bf16<true><<<dim3(M / 128, E / 128), 256, 0, stream>>>(
        attn, wpT, b_proj, (float*)d_out, M, E, E);
}
